// CrossFormer_31447750541455
// MI455X (gfx1250) — matrix operand order repaired and checked by reading, not compiled, not run on hardware
//
#include <hip/hip_runtime.h>
#include <hip/hip_bf16.h>

// ---------------------------------------------------------------------------
// CDNA5 (gfx1250) CrossFormer forward.
// Core = bf16 WMMA GEMM (v_wmma_f32_16x16x32_bf16), wave32, 8 waves/block,
// wave tile 32x64 (8 WMMA per k-step), block tile 128x128x32.
// Global->LDS staging is software-pipelined and double-buffered:
//   A tile: GLOBAL_LOAD_ASYNC_TO_LDS_B128 (ASYNCcnt)
//   B tile (TB=0): TENSOR_LOAD_TO_LDS via TDM descriptor (TENSORcnt)
// ---------------------------------------------------------------------------

typedef __attribute__((ext_vector_type(16))) __bf16        v16bf;
typedef __attribute__((ext_vector_type(8)))  float         v8f;
typedef __attribute__((ext_vector_type(4)))  unsigned int  v4u;
typedef __attribute__((ext_vector_type(8)))  int           v8i;
typedef __attribute__((ext_vector_type(4)))  int           v4i;

__device__ __forceinline__ unsigned short f2bf(float f) {
  unsigned int u = __builtin_bit_cast(unsigned int, f);
  unsigned int r = 0x7FFFu + ((u >> 16) & 1u);   // round-to-nearest-even
  return (unsigned short)((u + r) >> 16);
}
__device__ __forceinline__ unsigned pack2(float lo, float hi) {
  return (unsigned)f2bf(lo) | ((unsigned)f2bf(hi) << 16);
}

union FragU { v4u u[2]; v16bf h; };

// async 16B global -> LDS copy (per lane), tracked by ASYNCcnt
__device__ __forceinline__ void async_b128(unsigned lds_off, const float* gaddr) {
  asm volatile("global_load_async_to_lds_b128 %0, %1, off"
               :: "v"(lds_off), "v"(gaddr) : "memory");
}
__device__ __forceinline__ void wait_async(int n) {
  if (n == 0)      asm volatile("s_wait_asynccnt 0x0" ::: "memory");
  else if (n == 4) asm volatile("s_wait_asynccnt 0x4" ::: "memory");
  else             asm volatile("s_wait_asynccnt 0x8" ::: "memory");
}
__device__ __forceinline__ unsigned lds_off_of(const void* p) {
  return (unsigned)(unsigned long long)p;   // low 32 bits = LDS byte offset
}

// TDM: one whole 2-D f32 tile  global -> LDS  (D# per ISA ch.8; TENSORcnt)
// 6-arg builtin flavor: (v4u g0, v8i g1, v4i g2, v4i g3, v8i g4, i32 cpol)
__device__ __forceinline__ void tdm_load_2d(unsigned lds_addr, const void* gptr,
                                            unsigned tile0, unsigned tile1,
                                            unsigned long long stride0) {
  unsigned long long ga = (unsigned long long)gptr;
  v4u g0;
  g0[0] = 1u;                                       // count=1 (valid, user mode)
  g0[1] = lds_addr;                                 // lds_addr
  g0[2] = (unsigned)(ga & 0xFFFFFFFFu);             // global_addr[31:0]
  g0[3] = (unsigned)((ga >> 32) & 0x1FFFFFFu)       // global_addr[56:32]
        | (2u << 30);                               // type = 2 ("image")
  v8i g1;
  g1[0] = (int)(2u << 16);                          // wg_mask=0, data_size=4B
  g1[1] = (int)((tile0 & 0xFFFFu) << 16);           // tensor_dim0[15:0]
  g1[2] = (int)(((tile0 >> 16) & 0xFFFFu) | ((tile1 & 0xFFFFu) << 16)); // dim0 hi | dim1 lo
  g1[3] = (int)(((tile1 >> 16) & 0xFFFFu) | ((tile0 & 0xFFFFu) << 16)); // dim1 hi | tile_dim0
  g1[4] = (int)(tile1 & 0xFFFFu);                   // tile_dim1 (tile_dim2 = 0)
  g1[5] = (int)(stride0 & 0xFFFFFFFFu);             // tensor_dim0_stride[31:0]
  g1[6] = (int)((stride0 >> 32) & 0xFFFFu);         // stride hi (dim1_stride = 0)
  g1[7] = 0;
  v4i z4 = { 0, 0, 0, 0 };
  v8i z8 = { 0, 0, 0, 0, 0, 0, 0, 0 };
  __builtin_amdgcn_tensor_load_to_lds(g0, g1, z4, z4, z8, 0);
}

// ---------------------------------------------------------------------------
// Generic batched GEMM:  C = act(alpha * op(A) @ op(B) + bias) + res
//   op(A)(m,k) = TA ? A[k*lda+m] : A[m*lda+k]
//   op(B)(k,n) = TB ? B[n*ldb+k] : B[k*ldb+n]
// biasMode: -1 none, 0 bias[m], 1 bias[n].  act: 0 none,1 gelu,2 silu,3 relu.
// ---------------------------------------------------------------------------
#define BM 128
#define BN 128
#define BK 32
#define LPAD 40   // ushorts per LDS row: 80B rows, 16B-aligned chunks

template <int TA, int TB>
__global__ __launch_bounds__(256)
void k_gemm(const float* __restrict__ A, const float* __restrict__ B,
            float* __restrict__ C, const float* __restrict__ bias,
            const float* __restrict__ res,
            int M, int N, int K, int lda, int ldb, int ldc,
            int biasMode, int act, float alpha,
            long long sAo, long long sAi, int nAi,
            long long sBo, long long sBi, int nBi,
            long long sCo, long long sCi, int nCi)
{
  __shared__ float sA[2][BM * BK];              // f32 staging, double buffered
  __shared__ float sB[2][BK * BN];
  __shared__ unsigned short lA[BM][LPAD];       // packed bf16 operands
  __shared__ unsigned short lB[BN][LPAD];

  int z = blockIdx.z;
  A += (long long)(z / nAi) * sAo + (long long)(z % nAi) * sAi;
  B += (long long)(z / nBi) * sBo + (long long)(z % nBi) * sBi;
  long long coff = (long long)(z / nCi) * sCo + (long long)(z % nCi) * sCi;
  C += coff;
  if (res) res += coff;

  int m0 = blockIdx.y * BM;
  int n0 = blockIdx.x * BN;
  int t    = threadIdx.x;
  int lane = t & 31, wid = t >> 5;
  int wrow = wid & 3, wcol = wid >> 2;          // 4 x 2 wave grid
  int lr   = lane & 15, lh = lane >> 4;

  const unsigned sA_lds = lds_off_of(&sA[0][0]);
  const unsigned sB_lds = lds_off_of(&sB[0][0]);
  const bool fastmn = (m0 + BM <= M) && (n0 + BN <= N);
  const int  ntile  = (K + BK - 1) / BK;
  const int  ASY    = (TB == 0) ? 4 : 8;        // async instrs per tile per wave

  v8f acc[2][4];
  #pragma unroll
  for (int i = 0; i < 2; i++)
    #pragma unroll
    for (int j = 0; j < 4; j++)
      #pragma unroll
      for (int e = 0; e < 8; e++) acc[i][j][e] = 0.f;

  auto fastp = [&](int p) -> bool { return fastmn && (p * BK + BK <= K); };

  // issue async/TDM copies of tile p into staging buffer `buf`
  auto issue_tile = [&](int p, int buf) {
    const int kk = p * BK;
    const unsigned aoff = sA_lds + (unsigned)buf * (BM * BK * 4u);
    #pragma unroll
    for (int r = 0; r < 4; r++) {
      int id = r * 256 + t;
      if (TA == 0) {
        int row = id >> 3, seg = (id & 7) << 2;
        async_b128(aoff + (unsigned)(row * BK + seg) * 4u,
                   A + (long long)(m0 + row) * lda + kk + seg);
      } else {
        int kr = id >> 5, seg = (id & 31) << 2;
        async_b128(aoff + (unsigned)(kr * BM + seg) * 4u,
                   A + (long long)(kk + kr) * lda + m0 + seg);
      }
    }
    const unsigned boff = sB_lds + (unsigned)buf * (BK * BN * 4u);
    if (TB == 0) {
      if (wid == 0)                                // one TDM per tile
        tdm_load_2d(boff, B + (long long)kk * ldb + n0, BN, BK,
                    (unsigned long long)ldb);
    } else {
      #pragma unroll
      for (int r = 0; r < 4; r++) {
        int id = r * 256 + t;
        int nr = id >> 3, seg = (id & 7) << 2;
        async_b128(boff + (unsigned)(nr * BK + seg) * 4u,
                   B + (long long)(n0 + nr) * ldb + kk + seg);
      }
    }
  };

  // guarded scalar staging for edge tiles (clamped addr + select-zero)
  auto stage_slow = [&](int p, int buf) {
    const int kk = p * BK;
    int row = t >> 1, kb = (t & 1) << 4;
    {
      int m = m0 + row;
      bool mok = (m < M);
      int mc = mok ? m : (M - 1);
      #pragma unroll
      for (int i = 0; i < 16; i++) {
        int k = kk + kb + i;
        bool ok = mok && (k < K);
        int kc = (k < K) ? k : (K - 1);
        float v = TA ? A[(long long)kc * lda + mc] : A[(long long)mc * lda + kc];
        v = ok ? v : 0.f;
        if (TA) sA[buf][(kb + i) * BM + row] = v; else sA[buf][row * BK + kb + i] = v;
      }
    }
    {
      int n = n0 + row;
      bool nok = (n < N);
      int nc = nok ? n : (N - 1);
      #pragma unroll
      for (int i = 0; i < 16; i++) {
        int k = kk + kb + i;
        bool ok = nok && (k < K);
        int kc = (k < K) ? k : (K - 1);
        float v = TB ? B[(long long)nc * ldb + kc] : B[(long long)kc * ldb + nc];
        v = ok ? v : 0.f;
        if (TB) sB[buf][row * BK + kb + i] = v; else sB[buf][(kb + i) * BN + row] = v;
      }
    }
  };

  if (ntile > 0 && fastp(0)) issue_tile(0, 0);

  for (int p = 0; p < ntile; p++) {
    const int  cur = p & 1;
    const bool fp  = fastp(p);
    if (!fp) stage_slow(p, cur);
    const bool nf = (p + 1 < ntile) && fastp(p + 1);
    if (nf) issue_tile(p + 1, cur ^ 1);          // prefetch next tile
    if (fp) {                                    // drain current tile only
      wait_async(nf ? ASY : 0);
      if (TB == 0 && wid == 0) {
        if (nf) __builtin_amdgcn_s_wait_tensorcnt((short)1);
        else    __builtin_amdgcn_s_wait_tensorcnt((short)0);
      }
    }
    __syncthreads();

    // ---- f32 staging -> packed bf16 operand tiles (128-bit LDS stores) ----
    {
      int row = t >> 1, kb = (t & 1) << 4;
      #pragma unroll
      for (int half = 0; half < 2; half++) {
        v4u pv;
        #pragma unroll
        for (int j = 0; j < 4; j++) {
          int k = kb + half * 8 + 2 * j;
          float lo = TA ? sA[cur][k * BM + row]       : sA[cur][row * BK + k];
          float hi = TA ? sA[cur][(k + 1) * BM + row] : sA[cur][row * BK + k + 1];
          pv[j] = pack2(lo, hi);
        }
        *(v4u*)&lA[row][kb + half * 8] = pv;
      }
      #pragma unroll
      for (int half = 0; half < 2; half++) {
        v4u pv;
        #pragma unroll
        for (int j = 0; j < 4; j++) {
          int k = kb + half * 8 + 2 * j;
          float lo = TB ? sB[cur][row * BK + k]       : sB[cur][k * BN + row];
          float hi = TB ? sB[cur][row * BK + k + 1]   : sB[cur][(k + 1) * BN + row];
          pv[j] = pack2(lo, hi);
        }
        *(v4u*)&lB[row][kb + half * 8] = pv;
      }
    }
    __syncthreads();

    // ---- fragments (ISA VGPR layouts) + 8 WMMA ----
    v16bf af[2], bq[4];
    #pragma unroll
    for (int mt = 0; mt < 2; mt++) {
      // A 16x32 bf16: lane<16 holds K {0..7,16..23}; lane>=16 {8..15,24..31}
      FragU f;
      const unsigned short* p2 = &lA[wrow * 32 + mt * 16 + lr][lh * 8];
      f.u[0] = *(const v4u*)(p2);
      f.u[1] = *(const v4u*)(p2 + 16);
      af[mt] = f.h;
    }
    #pragma unroll
    for (int nt = 0; nt < 4; nt++) {
      // B 32x16 bf16: col = lane&15; lane<16 holds K {0..7,16..23}; lane>=16 {8..15,24..31}
      FragU f;
      const unsigned short* p2 = &lB[wcol * 64 + nt * 16 + lr][lh * 8];
      f.u[0] = *(const v4u*)(p2);
      f.u[1] = *(const v4u*)(p2 + 16);
      bq[nt] = f.h;
    }
    #pragma unroll
    for (int mt = 0; mt < 2; mt++)
      #pragma unroll
      for (int nt = 0; nt < 4; nt++)
        acc[mt][nt] = __builtin_amdgcn_wmma_f32_16x16x32_bf16(
            false, af[mt], false, bq[nt], (short)0, acc[mt][nt], false, false);
    __syncthreads();
  }

  // ---- fused epilogue ----
  #pragma unroll
  for (int mt = 0; mt < 2; mt++)
    #pragma unroll
    for (int nt = 0; nt < 4; nt++)
      #pragma unroll
      for (int r = 0; r < 8; r++) {
        int m = m0 + wrow * 32 + mt * 16 + lh * 8 + r;
        int n = n0 + wcol * 64 + nt * 16 + lr;
        if (m < M && n < N) {
          float v = acc[mt][nt][r] * alpha;
          if (biasMode == 0) v += bias[m];
          else if (biasMode == 1) v += bias[n];
          if (act == 1)      v = 0.5f * v * (1.f + erff(v * 0.70710678f));
          else if (act == 2) v = v / (1.f + expf(-v));
          else if (act == 3) v = fmaxf(v, 0.f);
          long long ci = (long long)m * ldc + n;
          if (res) v += res[ci];
          C[ci] = v;
        }
      }
}

// ---------------------------------------------------------------------------
// Channel layernorm over C at each spatial position (layout [C, HW]).
// ---------------------------------------------------------------------------
__global__ void k_chan_ln(const float* __restrict__ x, const float* __restrict__ g,
                          const float* __restrict__ b, float* __restrict__ y,
                          int C, int HW)
{
  int p = blockIdx.x * blockDim.x + threadIdx.x;
  if (p >= HW) return;
  float mu = 0.f;
  for (int c = 0; c < C; c++) mu += x[(long long)c * HW + p];
  mu /= C;
  float var = 0.f;
  for (int c = 0; c < C; c++) { float d = x[(long long)c * HW + p] - mu; var += d * d; }
  var /= C;
  float rstd = rsqrtf(var + 1e-5f);
  for (int c = 0; c < C; c++)
    y[(long long)c * HW + p] = (x[(long long)c * HW + p] - mu) * rstd * g[c] + b[c];
}

// ---------------------------------------------------------------------------
// Cube (patch) embedding conv: x[Cin,2,Hin,Win] stride (2,2,2) -> y[64,Hin/2,Win/2]
// ---------------------------------------------------------------------------
__global__ void k_cube_conv(const float* __restrict__ x, const float* __restrict__ w,
                            const float* __restrict__ b, float* __restrict__ y,
                            int Cin, int Hin, int Win, int Cout)
{
  int Ho = Hin / 2, Wo = Win / 2;
  long long tot = (long long)Cout * Ho * Wo;
  long long i = (long long)blockIdx.x * blockDim.x + threadIdx.x;
  if (i >= tot) return;
  int wo = (int)(i % Wo);
  int ho = (int)((i / Wo) % Ho);
  int o  = (int)(i / ((long long)Wo * Ho));
  float acc = b[o];
  for (int c = 0; c < Cin; c++)
    for (int f = 0; f < 2; f++)
      for (int p = 0; p < 2; p++)
        for (int q = 0; q < 2; q++)
          acc += x[(((long long)c * 2 + f) * Hin + (2 * ho + p)) * Win + (2 * wo + q)] *
                 w[((((long long)o * Cin + c) * 2 + f) * 2 + p) * 2 + q];
  y[i] = acc;
}

// ---------------------------------------------------------------------------
// Direct conv (CEL + 3x3 up-path), writes channel slice [coff, coff+Cout).
// ---------------------------------------------------------------------------
__global__ void k_conv2d(const float* __restrict__ x, const float* __restrict__ w,
                         const float* __restrict__ b, float* __restrict__ y,
                         int Cin, int Hin, int Win, int Cout, int Ho, int Wo,
                         int ks, int st, int pad, int coff, int Ctot)
{
  (void)Ctot;
  long long tot = (long long)Cout * Ho * Wo;
  long long i = (long long)blockIdx.x * blockDim.x + threadIdx.x;
  if (i >= tot) return;
  int wo = (int)(i % Wo);
  int ho = (int)((i / Wo) % Ho);
  int o  = (int)(i / ((long long)Wo * Ho));
  float acc = b ? b[o] : 0.f;
  int h0 = ho * st - pad, w0 = wo * st - pad;
  for (int c = 0; c < Cin; c++)
    for (int kh = 0; kh < ks; kh++) {
      int hh = h0 + kh; if (hh < 0 || hh >= Hin) continue;
      for (int kw = 0; kw < ks; kw++) {
        int ww = w0 + kw; if (ww < 0 || ww >= Win) continue;
        acc += x[((long long)c * Hin + hh) * Win + ww] *
               w[(((long long)o * Cin + c) * ks + kh) * ks + kw];
      }
    }
  y[((long long)(coff + o) * Ho + ho) * Wo + wo] = acc;
}

// ---------------------------------------------------------------------------
// Window gather [D,H,W] -> [nwin, D, wsz*wsz] and inverse scatter-add.
// ---------------------------------------------------------------------------
__global__ void k_wgather(const float* __restrict__ x, float* __restrict__ y,
                          int D, int H, int W, int wsz, int isLong)
{
  int GH = H / wsz, GW = W / wsz, T = wsz * wsz;
  long long tot = (long long)D * H * W;
  long long o = (long long)blockIdx.x * blockDim.x + threadIdx.x;
  if (o >= tot) return;
  int tk = (int)(o % T);
  int d  = (int)((o / T) % D);
  int wi = (int)(o / ((long long)T * D));
  int ti = tk / wsz, tj = tk % wsz;
  int gh = wi / GW, gw = wi % GW;
  int h, w;
  if (isLong) { h = ti * GH + gh; w = tj * GW + gw; }
  else        { h = gh * wsz + ti; w = gw * wsz + tj; }
  y[o] = x[((long long)d * H + h) * W + w];
}

__global__ void k_wscatter_add(const float* __restrict__ y, float* __restrict__ x,
                               int D, int H, int W, int wsz, int isLong)
{
  int GH = H / wsz, GW = W / wsz, T = wsz * wsz;
  long long tot = (long long)D * H * W;
  long long o = (long long)blockIdx.x * blockDim.x + threadIdx.x;
  if (o >= tot) return;
  int tk = (int)(o % T);
  int d  = (int)((o / T) % D);
  int wi = (int)(o / ((long long)T * D));
  int ti = tk / wsz, tj = tk % wsz;
  int gh = wi / GW, gw = wi % GW;
  int h, w;
  if (isLong) { h = ti * GH + gh; w = tj * GW + gw; }
  else        { h = gh * wsz + ti; w = gw * wsz + tj; }
  x[((long long)d * H + h) * W + w] += y[o];
}

// ---------------------------------------------------------------------------
// Relative-position tables (device-side numpy replacement).
// ---------------------------------------------------------------------------
__global__ void k_relpos(int* __restrict__ idx, float* __restrict__ mg, int wsz)
{
  int T = wsz * wsz, P2 = 2 * wsz + 1;
  int i = blockIdx.x * blockDim.x + threadIdx.x;
  if (i < T * T) {
    int a = i / T, b = i % T;
    int ry = (a / wsz) - (b / wsz) + wsz - 1;
    int rx = (a % wsz) - (b % wsz) + wsz - 1;
    idx[i] = ry * (2 * wsz - 1) + rx;
  }
  if (i < P2 * P2) {
    mg[i * 2 + 0] = (float)(i / P2 - wsz);
    mg[i * 2 + 1] = (float)(i % P2 - wsz);
  }
}

// ---------------------------------------------------------------------------
// Dynamic-position-bias MLP: one block per relative position, h <= 128.
// ---------------------------------------------------------------------------
__global__ __launch_bounds__(128)
void k_dpb(const float* __restrict__ mg,
           const float* w1, const float* b1, const float* g1, const float* e1,
           const float* w2, const float* b2, const float* g2, const float* e2,
           const float* w3, const float* b3, const float* g3, const float* e3,
           const float* w4, const float* b4,
           float* __restrict__ bt, int h)
{
  __shared__ float hv[128];
  __shared__ float red[128];
  int r = blockIdx.x, tid = threadIdx.x;
  float x0 = mg[r * 2 + 0], x1 = mg[r * 2 + 1];
  const float* G[3] = { g1, g2, g3 };
  const float* E[3] = { e1, e2, e3 };
  const float* Wn[2] = { w2, w3 };
  const float* Bn[2] = { b2, b3 };

  float a = 0.f;
  if (tid < h) a = w1[tid * 2 + 0] * x0 + w1[tid * 2 + 1] * x1 + b1[tid];

  for (int layer = 0; layer < 3; layer++) {
    red[tid] = (tid < h) ? a : 0.f; __syncthreads();
    for (int s = 64; s > 0; s >>= 1) { if (tid < s) red[tid] += red[tid + s]; __syncthreads(); }
    float mu = red[0] / h; __syncthreads();
    float d = (tid < h) ? a - mu : 0.f;
    red[tid] = d * d; __syncthreads();
    for (int s = 64; s > 0; s >>= 1) { if (tid < s) red[tid] += red[tid + s]; __syncthreads(); }
    float rstd = rsqrtf(red[0] / h + 1e-5f); __syncthreads();
    float o = 0.f;
    if (tid < h) o = fmaxf(d * rstd * G[layer][tid] + E[layer][tid], 0.f);
    hv[tid] = o; __syncthreads();
    if (layer < 2) {
      float s = 0.f;
      if (tid < h) {
        const float* Wm = Wn[layer];
        for (int j = 0; j < h; j++) s += Wm[tid * h + j] * hv[j];
        s += Bn[layer][tid];
      }
      a = s; __syncthreads();
    }
  }
  red[tid] = (tid < h) ? w4[tid] * hv[tid] : 0.f; __syncthreads();
  for (int s = 64; s > 0; s >>= 1) { if (tid < s) red[tid] += red[tid + s]; __syncthreads(); }
  if (tid == 0) bt[r] = red[0] + b4[0];
}

// ---------------------------------------------------------------------------
// Row softmax with gathered relative-position bias.
// ---------------------------------------------------------------------------
__global__ void k_softmax(float* __restrict__ P, const float* __restrict__ bt,
                          const int* __restrict__ idx, int T, long long rows)
{
  long long rI = (long long)blockIdx.x * blockDim.x + threadIdx.x;
  if (rI >= rows) return;
  int i = (int)(rI % T);
  float* row = P + rI * (long long)T;
  const int* ir = idx + (long long)i * T;
  float mx = -1e30f;
  for (int j = 0; j < T; j++) { float v = row[j] + bt[ir[j]]; row[j] = v; if (v > mx) mx = v; }
  float s = 0.f;
  for (int j = 0; j < T; j++) { float e = expf(row[j] - mx); row[j] = e; s += e; }
  float inv = 1.f / s;
  for (int j = 0; j < T; j++) row[j] *= inv;
}

// ---------------------------------------------------------------------------
// 2x2 stride-2 transposed conv (einsum bcij,copq->boipjq).
// ---------------------------------------------------------------------------
__global__ void k_convt(const float* __restrict__ y, const float* __restrict__ w,
                        const float* __restrict__ b, float* __restrict__ up,
                        int Cin, int H, int W, int Cout)
{
  long long tot = (long long)Cout * 2 * H * 2 * W;
  long long i = (long long)blockIdx.x * blockDim.x + threadIdx.x;
  if (i >= tot) return;
  int W2 = 2 * W, H2 = 2 * H;
  int ow = (int)(i % W2);
  int oh = (int)((i / W2) % H2);
  int o  = (int)(i / ((long long)W2 * H2));
  int ih = oh >> 1, p = oh & 1, iw = ow >> 1, q = ow & 1;
  float acc = b[o];
  for (int c = 0; c < Cin; c++)
    acc += y[((long long)c * H + ih) * W + iw] *
           w[(((long long)c * Cout + o) * 2 + p) * 2 + q];
  up[i] = acc;
}

// ---------------------------------------------------------------------------
// GroupNorm + SiLU, one block per group.
// ---------------------------------------------------------------------------
__global__ __launch_bounds__(256)
void k_gn_silu(const float* __restrict__ x, const float* __restrict__ g,
               const float* __restrict__ b, float* __restrict__ y,
               int C, int G, int HW)
{
  __shared__ float red[256];
  int grp = blockIdx.x, tid = threadIdx.x;
  int Cg = C / G;
  long long base = (long long)grp * Cg * HW;
  long long n = (long long)Cg * HW;
  float s = 0.f;
  for (long long i = tid; i < n; i += 256) s += x[base + i];
  red[tid] = s; __syncthreads();
  for (int st = 128; st > 0; st >>= 1) { if (tid < st) red[tid] += red[tid + st]; __syncthreads(); }
  float mu = red[0] / (float)n; __syncthreads();
  s = 0.f;
  for (long long i = tid; i < n; i += 256) { float d = x[base + i] - mu; s += d * d; }
  red[tid] = s; __syncthreads();
  for (int st = 128; st > 0; st >>= 1) { if (tid < st) red[tid] += red[tid + st]; __syncthreads(); }
  float rstd = rsqrtf(red[0] / (float)n + 1e-5f);
  for (long long i = tid; i < n; i += 256) {
    int c = grp * Cg + (int)(i / HW);
    float v = (x[base + i] - mu) * rstd * g[c] + b[c];
    y[base + i] = v / (1.f + expf(-v));
  }
}

__global__ void k_add(float* __restrict__ a, const float* __restrict__ b, long long n)
{
  long long i = (long long)blockIdx.x * blockDim.x + threadIdx.x;
  if (i < n) a[i] += b[i];
}

// ===========================================================================
// Host orchestration
// ===========================================================================
extern "C" void kernel_launch(void* const* d_in, const int* in_sizes, int n_in,
                              void* d_out, int out_size, void* d_ws, size_t ws_size,
                              hipStream_t stream)
{
  (void)in_sizes; (void)n_in; (void)out_size; (void)ws_size;

  static const int DIMS_[4]   = { 64, 128, 256, 512 };
  static const int DEPTHS_[4] = { 2, 2, 8, 2 };
  static const int GWSZ_[4]   = { 10, 5, 2, 1 };
  static const int CEL_ST_[4] = { 4, 2, 2, 2 };
  static const int K0_[4] = { 4, 8, 16, 32 };
  static const int K1_[2] = { 2, 4 };
  const int IN_CH = 4 * 15 + 7;   // 67

  // ---- walk the flattened parameter dict (insertion order) ----
  int ip = 0;
  auto nf = [&]() -> const float* { return (const float*)d_in[ip++]; };
  const float* X = nf();
  const float *cw = nf(), *cb = nf(), *clnw = nf(), *clnb = nf();

  struct Dpb { const float *w1,*b1,*g1,*e1,*w2,*b2,*g2,*e2,*w3,*b3,*g3,*e3,*w4,*b4; };
  struct Att { const float *ng,*nb,*qkv,*ow,*ob; Dpb dpb; };
  struct Ffp { const float *ng,*nb,*w1,*b1,*w2,*b2; };
  struct Blk { Att sa; Ffp sf; Att la; Ffp lf; };
  struct Stage { const float* celw[4]; const float* celb[4]; int nk; Blk blk[8]; };
  Stage st[4];

  auto rdDpb = [&](Dpb& d) {
    d.w1=nf(); d.b1=nf(); d.g1=nf(); d.e1=nf();
    d.w2=nf(); d.b2=nf(); d.g2=nf(); d.e2=nf();
    d.w3=nf(); d.b3=nf(); d.g3=nf(); d.e3=nf();
    d.w4=nf(); d.b4=nf();
  };
  auto rdAtt = [&](Att& a) { a.ng=nf(); a.nb=nf(); a.qkv=nf(); a.ow=nf(); a.ob=nf(); rdDpb(a.dpb); };
  auto rdFf  = [&](Ffp& f) { f.ng=nf(); f.nb=nf(); f.w1=nf(); f.b1=nf(); f.w2=nf(); f.b2=nf(); };
  for (int si = 0; si < 4; si++) {
    st[si].nk = (si == 0) ? 4 : 2;
    for (int k = 0; k < st[si].nk; k++) { st[si].celw[k] = nf(); st[si].celb[k] = nf(); }
    for (int bi = 0; bi < DEPTHS_[si]; bi++) {
      rdAtt(st[si].blk[bi].sa); rdFf(st[si].blk[bi].sf);
      rdAtt(st[si].blk[bi].la); rdFf(st[si].blk[bi].lf);
    }
  }
  const float *ctw = nf(), *ctb = nf(), *c1w = nf(), *c1b = nf(),
              *g1w = nf(), *g1b = nf(), *c2w = nf(), *c2b = nf(),
              *g2w = nf(), *g2b = nf(), *fw = nf(), *fb = nf();

  // ---- workspace bump allocator ----
  char* wsp = (char*)d_ws;
  size_t wo = 0;
  auto alloc = [&](size_t bytes) -> void* {
    void* p = wsp + wo; wo += (bytes + 255) & ~(size_t)255; return p;
  };
  float* Y    = (float*)alloc(sizeof(float) * 64ll * 320 * 640);
  float* Tb   = (float*)alloc(sizeof(float) * 64ll * 320 * 640);
  float* XN   = (float*)alloc(sizeof(float) * 1100000);
  float* XW   = (float*)alloc(sizeof(float) * 1100000);
  float* QKV  = (float*)alloc(sizeof(float) * 3300000);
  float* Pbuf = (float*)alloc(sizeof(float) * 3000000);
  float* AO   = (float*)alloc(sizeof(float) * 1100000);
  float* PR   = (float*)alloc(sizeof(float) * 1100000);
  float* FH   = (float*)alloc(sizeof(float) * 4200000);
  int*   IDX  = (int*)  alloc(sizeof(int)   * 10000);
  float* MG   = (float*)alloc(sizeof(float) * 1024);
  float* BT   = (float*)alloc(sizeof(float) * 512);

  auto gemm = [&](const float* A, const float* B, float* C,
                  const float* bias, const float* res,
                  int M, int N, int K, int lda, int ldb, int ldc,
                  int tA, int tB, int bm, int act, float alpha, int batch,
                  long long sAo, long long sAi, int nAi,
                  long long sBo, long long sBi, int nBi,
                  long long sCo, long long sCi, int nCi) {
    dim3 grid((N + BN - 1) / BN, (M + BM - 1) / BM, batch);
    if (nAi < 1) nAi = 1;
    if (nBi < 1) nBi = 1;
    if (nCi < 1) nCi = 1;
    if (!tA && !tB)
      k_gemm<0,0><<<grid, 256, 0, stream>>>(A, B, C, bias, res, M, N, K, lda, ldb, ldc,
          bm, act, alpha, sAo, sAi, nAi, sBo, sBi, nBi, sCo, sCi, nCi);
    else if (tA && !tB)
      k_gemm<1,0><<<grid, 256, 0, stream>>>(A, B, C, bias, res, M, N, K, lda, ldb, ldc,
          bm, act, alpha, sAo, sAi, nAi, sBo, sBi, nBi, sCo, sCi, nCi);
    else if (!tA && tB)
      k_gemm<0,1><<<grid, 256, 0, stream>>>(A, B, C, bias, res, M, N, K, lda, ldb, ldc,
          bm, act, alpha, sAo, sAi, nAi, sBo, sBi, nBi, sCo, sCi, nCi);
    else
      k_gemm<1,1><<<grid, 256, 0, stream>>>(A, B, C, bias, res, M, N, K, lda, ldb, ldc,
          bm, act, alpha, sAo, sAi, nAi, sBo, sBi, nBi, sCo, sCi, nCi);
  };

  // ---- cube embedding + LN ----
  {
    long long tot = 64ll * 320 * 640;
    k_cube_conv<<<(unsigned)((tot + 255) / 256), 256, 0, stream>>>(
        X, cw, cb, Tb, IN_CH, 640, 1280, 64);
    k_chan_ln<<<(320 * 640 + 255) / 256, 256, 0, stream>>>(
        Tb, clnw, clnb, Y, 64, 320 * 640);
  }

  // ---- stages ----
  int H = 320, W = 640, Din = 64;
  for (int si = 0; si < 4; si++) {
    int D = DIMS_[si], stx = CEL_ST_[si];
    int Ho = H / stx, Wo = W / stx;

    // CEL: multi-kernel strided conv, channel-concatenated into Tb
    int coff = 0, rem = D;
    for (int ki = 0; ki < st[si].nk; ki++) {
      int ks = (si == 0) ? K0_[ki] : K1_[ki];
      int dc = (ki < st[si].nk - 1) ? (D >> (ki + 1)) : rem;
      rem -= dc;
      int pad = (ks - stx) / 2;
      long long tot = (long long)dc * Ho * Wo;
      k_conv2d<<<(unsigned)((tot + 255) / 256), 256, 0, stream>>>(
          Y, st[si].celw[ki], st[si].celb[ki], Tb,
          Din, H, W, dc, Ho, Wo, ks, stx, pad, coff, D);
      coff += dc;
    }
    { float* t = Y; Y = Tb; Tb = t; }
    H = Ho; W = Wo; Din = D;
    int HW = H * W, heads = D / 32;

    auto attn = [&](const Att& ap, int wsz, int isLong) {
      int T = wsz * wsz, GH = H / wsz, GW = W / wsz, nwin = GH * GW;
      long long tot = (long long)D * HW;
      k_chan_ln<<<(HW + 255) / 256, 256, 0, stream>>>(Y, ap.ng, ap.nb, XN, D, HW);
      k_wgather<<<(unsigned)((tot + 255) / 256), 256, 0, stream>>>(
          XN, XW, D, H, W, wsz, isLong);
      // qkv: [3D x T] = Wqkv[3D x D] @ xw[D x T], batched over windows
      gemm(ap.qkv, XW, QKV, nullptr, nullptr,
           3 * D, T, D, D, T, T, 0, 0, -1, 0, 1.f, nwin,
           0, 0, 1, (long long)D * T, 0, 1, 3ll * D * T, 0, 1);
      // relative-position bias
      int P2 = 2 * wsz + 1;
      int mwork = (T * T > P2 * P2) ? T * T : P2 * P2;
      k_relpos<<<(mwork + 255) / 256, 256, 0, stream>>>(IDX, MG, wsz);
      k_dpb<<<P2 * P2, 128, 0, stream>>>(MG,
          ap.dpb.w1, ap.dpb.b1, ap.dpb.g1, ap.dpb.e1,
          ap.dpb.w2, ap.dpb.b2, ap.dpb.g2, ap.dpb.e2,
          ap.dpb.w3, ap.dpb.b3, ap.dpb.g3, ap.dpb.e3,
          ap.dpb.w4, ap.dpb.b4, BT, D / 4);
      // sim = (q^T k) * rsqrt(32), batched over (window, head)
      gemm(QKV, QKV + (long long)D * T, Pbuf, nullptr, nullptr,
           T, T, 32, T, T, T, 1, 0, -1, 0, 0.1767766953f, nwin * heads,
           3ll * D * T, 32ll * T, heads,
           3ll * D * T, 32ll * T, heads,
           (long long)heads * T * T, (long long)T * T, heads);
      long long rows = (long long)nwin * heads * T;
      k_softmax<<<(unsigned)((rows + 255) / 256), 256, 0, stream>>>(
          Pbuf, BT, IDX, T, rows);
      // out = v @ P^T  ->  [nwin, D, T]
      gemm(QKV + 2ll * D * T, Pbuf, AO, nullptr, nullptr,
           32, T, T, T, T, T, 0, 1, -1, 0, 1.f, nwin * heads,
           3ll * D * T, 32ll * T, heads,
           (long long)heads * T * T, (long long)T * T, heads,
           (long long)D * T, 32ll * T, heads);
      // proj: [D x T] = Wo[D x D] @ out + bias, batched over windows
      gemm(ap.ow, AO, PR, ap.ob, nullptr,
           D, T, D, D, T, T, 0, 0, 0, 0, 1.f, nwin,
           0, 0, 1, (long long)D * T, 0, 1, (long long)D * T, 0, 1);
      // residual scatter back into Y
      k_wscatter_add<<<(unsigned)((tot + 255) / 256), 256, 0, stream>>>(
          PR, Y, D, H, W, wsz, isLong);
    };

    auto ffn = [&](const Ffp& fp) {
      k_chan_ln<<<(HW + 255) / 256, 256, 0, stream>>>(Y, fp.ng, fp.nb, XN, D, HW);
      gemm(fp.w1, XN, FH, fp.b1, nullptr,
           4 * D, HW, D, D, HW, HW, 0, 0, 0, 1 /*gelu*/, 1.f, 1,
           0, 0, 1, 0, 0, 1, 0, 0, 1);
      gemm(fp.w2, FH, Y, fp.b2, Y /*residual, in place*/,
           D, HW, 4 * D, 4 * D, HW, HW, 0, 0, 0, 0, 1.f, 1,
           0, 0, 1, 0, 0, 1, 0, 0, 1);
    };

    for (int bi = 0; bi < DEPTHS_[si]; bi++) {
      Blk& bl = st[si].blk[bi];
      attn(bl.sa, 10,        0);
      ffn (bl.sf);
      attn(bl.la, GWSZ_[si], 1);
      ffn (bl.lf);
    }
  }

  // ---- up path ----  (Y = [512, 10, 20])
  {
    const int half = 256, Hs = 10, Ws = 20, HW2 = (2 * Hs) * (2 * Ws);
    float* UP = XN;
    float* Z1 = FH;
    float* Z2 = Tb;
    long long tot = (long long)half * HW2;
    k_convt<<<(unsigned)((tot + 255) / 256), 256, 0, stream>>>(
        Y, ctw, ctb, UP, 512, Hs, Ws, half);
    k_conv2d<<<(unsigned)((tot + 255) / 256), 256, 0, stream>>>(
        UP, c1w, c1b, Z1, half, 2 * Hs, 2 * Ws, half, 2 * Hs, 2 * Ws, 3, 1, 1, 0, half);
    k_gn_silu<<<64, 256, 0, stream>>>(Z1, g1w, g1b, Z2, half, 64, HW2);
    k_conv2d<<<(unsigned)((tot + 255) / 256), 256, 0, stream>>>(
        Z2, c2w, c2b, Z1, half, 2 * Hs, 2 * Ws, half, 2 * Hs, 2 * Ws, 3, 1, 1, 0, half);
    k_gn_silu<<<64, 256, 0, stream>>>(Z1, g2w, g2b, Z2, half, 64, HW2);
    k_add<<<(unsigned)((tot + 255) / 256), 256, 0, stream>>>(Z2, UP, tot);
    // final fc: out[HW2 x 67] = Z2^T[HW2 x 256] @ fw^T[256 x 67] + fb
    gemm(Z2, fw, (float*)d_out, fb, nullptr,
         HW2, IN_CH, half, HW2 /*lda (transA)*/, half /*ldb (transB)*/, IN_CH,
         1, 1, 1 /*bias per-n*/, 0, 1.f, 1,
         0, 0, 1, 0, 0, 1, 0, 0, 1);
  }
}
